// IFLM_Spatial_Channel_61306363183747
// MI455X (gfx1250) — matrix operand order repaired and checked by reading, not compiled, not run on hardware
//
#include <hip/hip_runtime.h>

typedef __attribute__((ext_vector_type(16))) __bf16 v16bf;
typedef __attribute__((ext_vector_type(8)))  __bf16 v8bf;
typedef __attribute__((ext_vector_type(8)))  float  v8f;
typedef unsigned int v4u __attribute__((ext_vector_type(4)));
typedef int          v8i __attribute__((ext_vector_type(8)));
typedef int          v4i __attribute__((ext_vector_type(4)));

#define HP   78      // 64 + 2*7 padded extent
#define PAD  7
#define CIN  512
#define COUT 128
#define ROW_ELEMS  (HP * CIN)        // 39936 bf16 per padded row
#define PIX_BYTES  1040              // 1024 B channels + 16 B TDM pad (bank-conflict-free)
#define LDS_BYTES  (3 * HP * PIX_BYTES)   // 243,360 B  (<320 KB WGP LDS)

// ---------------------------------------------------------------------------
// Pack x: NCHW f32 [2,512,64,64] -> zero-padded channels-last bf16 [2,78,78,512]
// ---------------------------------------------------------------------------
__global__ void pack_x_kernel(const float* __restrict__ x, __bf16* __restrict__ xp, int n) {
    int i = blockIdx.x * blockDim.x + threadIdx.x;
    if (i >= n) return;
    int c  = i & (CIN - 1);
    int t  = i >> 9;
    int wq = t % HP;
    int hq = (t / HP) % HP;
    int b  = t / (HP * HP);
    int h = hq - PAD, w = wq - PAD;
    float v = 0.0f;
    if ((unsigned)h < 64u && (unsigned)w < 64u)
        v = x[(((size_t)b * CIN + c) * 64 + h) * 64 + w];
    xp[i] = (__bf16)v;
}

// ---------------------------------------------------------------------------
// Pack weights: OIHW f32 [128,512,3,3] (W2 and W4) -> bf16 [conv][tap][o][c]
// ---------------------------------------------------------------------------
__global__ void pack_w_kernel(const float* __restrict__ W2, const float* __restrict__ W4,
                              __bf16* __restrict__ wp, int n) {
    int i = blockIdx.x * blockDim.x + threadIdx.x;
    if (i >= n) return;
    int c    = i & 511;
    int r    = i >> 9;
    int o    = r & 127;
    int q    = r >> 7;
    int tap  = q % 9;
    int conv = q / 9;
    const float* Wsrc = conv ? W4 : W2;
    wp[i] = (__bf16)Wsrc[(size_t)(o * 512 + c) * 9 + tap];
}

// ---------------------------------------------------------------------------
// Dilated conv (512->128, 3x3, d=3 or 7) + bias + ReLU -> per-row partial sums.
// One block (8 waves) = one (conv,b,row): 128 oc x 64 px.  The 3 input rows the
// row needs (234 KB) are DMA'd into LDS once via the Tensor Data Mover with a
// 16B/pixel pad so B-operand ds_load_b128's sweep all 64 banks.  Each wave then
// runs a 32x32 implicit-GEMM tile: 9 taps x 16 k-steps of V_WMMA_F32_16X16X32_BF16.
// ---------------------------------------------------------------------------
__global__ void __launch_bounds__(256) conv_wmma_kernel(
    const __bf16* __restrict__ xp, const __bf16* __restrict__ wp,
    const float* __restrict__ bias2, const float* __restrict__ bias4,
    float* __restrict__ part) {

    extern __shared__ char smem[];

    const int tid  = threadIdx.x;
    const int wave = tid >> 5;
    const int lane = tid & 31;

    const int h    = blockIdx.x & 63;
    const int b    = (blockIdx.x >> 6) & 1;
    const int conv = (blockIdx.x >> 7) & 1;        // 0: W2/d=3  1: W4/d=7
    const int d    = conv ? 7 : 3;

    const int wh   = wave & 1;                     // 32-pixel half-row
    const int oct  = (wave >> 1) & 3;              // 32-channel tile

    const int laneN = lane & 15;
    const int kbA   = (lane & 16) ? 8  : 0;        // ISA 16-bit A-matrix layout
    const int kbB   = (lane & 16) ? 8  : 0;        // ISA 16-bit B-matrix layout

    // ---- stage activations: rows (h+7-d, h+7, h+7+d), full padded width -----
    const size_t src_row0 = ((size_t)b * HP + (h + PAD - d)) * HP * CIN;  // elem idx

#if __has_builtin(__builtin_amdgcn_tensor_load_to_lds)
    if (wave == 0) {
        unsigned lds_base = (unsigned)(unsigned long long)(void*)smem;  // LDS byte offset
        unsigned long long ga = (unsigned long long)(const void*)(xp + src_row0);
        const unsigned stride0 = (unsigned)(d * ROW_ELEMS);             // row stride (elems)

        // D# group 0: count=1 | lds_addr | global_addr[56:0] | type=2
        v4u g0 = { 1u, lds_base, (unsigned)ga,
                   (unsigned)((ga >> 32) & 0x01FFFFFFull) | (2u << 30) };
        // D# group 1: data_size=2B, pad_enable, pad_interval=256dw, pad_amount=4dw,
        //             tensor_dim0=tile_dim0=39936, tensor_dim1=3, tile_dim1=3, dim0_stride
        v8i g1 = { (int)((1u << 16) | (1u << 20) | (7u << 22) | (3u << 25)),
                   (int)((ROW_ELEMS & 0xFFFFu) << 16),
                   (int)((ROW_ELEMS >> 16) | (3u << 16)),
                   (int)((ROW_ELEMS & 0xFFFFu) << 16),
                   (int)3,
                   (int)stride0,
                   0, 0 };
        v4i gz4 = { 0, 0, 0, 0 };
#if defined(__clang_major__) && (__clang_major__ >= 23)
        v8i gz8 = { 0, 0, 0, 0, 0, 0, 0, 0 };
        __builtin_amdgcn_tensor_load_to_lds(g0, g1, gz4, gz4, gz8, 0);
#else
        __builtin_amdgcn_tensor_load_to_lds(g0, g1, gz4, gz4, 0);
#endif
        __builtin_amdgcn_s_wait_tensorcnt(0);
    }
#else
    // Fallback: cooperative copy reproducing the same padded LDS layout.
    for (int q = tid; q < 3 * HP * 64; q += 256) {       // 16B chunks
        int p  = q >> 6;                                  // pixel index (row-major 3x78)
        int ch = q & 63;                                  // 16B chunk within pixel
        int r  = p / HP, wq = p % HP;
        const __bf16* src = xp + src_row0 + ((size_t)r * d * HP + wq) * CIN + ch * 8;
        *(v8bf*)(smem + (size_t)p * PIX_BYTES + ch * 16) = *(const v8bf*)src;
    }
#endif
    __syncthreads();

    // ---- main WMMA loops -----------------------------------------------------
    v8f acc00 = {}, acc01 = {}, acc10 = {}, acc11 = {};
    const __bf16* wconv = wp + (size_t)conv * 9 * COUT * CIN;

    for (int t = 0; t < 9; ++t) {
        const int dw = t % 3 - 1;
        const int r  = t / 3;                                   // LDS row (dh = r-1)
        const __bf16* arow0 = wconv + ((size_t)t * COUT + oct * 32 + laneN) * CIN;
        const __bf16* arow1 = arow0 + (size_t)16 * CIN;
        __builtin_prefetch(arow0, 0, 1);                        // global_prefetch_b8
        __builtin_prefetch(arow1, 0, 1);

        const int bcol = wh * 32 + laneN + PAD + dw * d;        // 0..77
        const char* brow0 = smem + ((size_t)(r * HP + bcol)) * PIX_BYTES;
        const char* brow1 = brow0 + (size_t)16 * PIX_BYTES;

        #pragma unroll
        for (int ks = 0; ks < 16; ++ks) {
            const int cb = ks * 32;
            // A: lane holds K = kbA..kbA+7 (v0..3) and 16+kbA..+7 (v4..7)
            v8bf a0lo = *(const v8bf*)(arow0 + cb + kbA);
            v8bf a0hi = *(const v8bf*)(arow0 + cb + 16 + kbA);
            v8bf a1lo = *(const v8bf*)(arow1 + cb + kbA);
            v8bf a1hi = *(const v8bf*)(arow1 + cb + 16 + kbA);
            v16bf a0 = __builtin_shufflevector(a0lo, a0hi, 0,1,2,3,4,5,6,7,8,9,10,11,12,13,14,15);
            v16bf a1 = __builtin_shufflevector(a1lo, a1hi, 0,1,2,3,4,5,6,7,8,9,10,11,12,13,14,15);
            // B from LDS: lane holds K = kbB..kbB+7 (v0..3) and 16+kbB..+7 (v4..7)
            const int co = (cb + kbB) * 2;
            v8bf b0lo = *(const v8bf*)(brow0 + co);
            v8bf b0hi = *(const v8bf*)(brow0 + co + 32);
            v8bf b1lo = *(const v8bf*)(brow1 + co);
            v8bf b1hi = *(const v8bf*)(brow1 + co + 32);
            v16bf b0 = __builtin_shufflevector(b0lo, b0hi, 0,1,2,3,4,5,6,7,8,9,10,11,12,13,14,15);
            v16bf b1 = __builtin_shufflevector(b1lo, b1hi, 0,1,2,3,4,5,6,7,8,9,10,11,12,13,14,15);

            acc00 = __builtin_amdgcn_wmma_f32_16x16x32_bf16(false, a0, false, b0, (short)0, acc00, false, false);
            acc01 = __builtin_amdgcn_wmma_f32_16x16x32_bf16(false, a0, false, b1, (short)0, acc01, false, false);
            acc10 = __builtin_amdgcn_wmma_f32_16x16x32_bf16(false, a1, false, b0, (short)0, acc10, false, false);
            acc11 = __builtin_amdgcn_wmma_f32_16x16x32_bf16(false, a1, false, b1, (short)0, acc11, false, false);
        }
    }

    // ---- epilogue: bias + ReLU, reduce the 16 N-lanes that share M ----------
    // C/D layout: VGPR v, lanes 0-15 -> M=v, lanes 16-31 -> M=v+8; N = lane&15.
    const float* bias = conv ? bias4 : bias2;
    const int hiM = (lane & 16) ? 8 : 0;
    const float* bp0 = bias + oct * 32 + hiM;
    const float* bp1 = bias + oct * 32 + 16 + hiM;
    const int pair = conv * 2 + b;
    const int col  = h * 2 + wh;

    float s0[8], s1[8];
    #pragma unroll
    for (int v = 0; v < 8; ++v) {
        float b0v = bp0[v], b1v = bp1[v];
        float t0 = fmaxf(acc00[v] + b0v, 0.0f) + fmaxf(acc01[v] + b0v, 0.0f);
        float t1 = fmaxf(acc10[v] + b1v, 0.0f) + fmaxf(acc11[v] + b1v, 0.0f);
        #pragma unroll
        for (int off = 1; off < 16; off <<= 1) {   // stays within each 16-lane half
            t0 += __shfl_xor(t0, off, 32);
            t1 += __shfl_xor(t1, off, 32);
        }
        s0[v] = t0; s1[v] = t1;
    }
    if (laneN == 0) {   // lanes 0 and 16 hold the reduced sums for their M set
        #pragma unroll
        for (int v = 0; v < 8; ++v) {
            int oc0 = oct * 32 + hiM + v;
            int oc1 = oct * 32 + 16 + hiM + v;
            part[((size_t)pair * COUT + oc0) * 128 + col] = s0[v];
            part[((size_t)pair * COUT + oc1) * 128 + col] = s1[v];
        }
    }
}

// ---------------------------------------------------------------------------
// Dense chain: m = GAP -> Wd(512x128) -> ccifl (Wcc+bcc, softmax of one logit
// == 1) -> Wd2(512x256); summed over the two conv branches.  ~1M MACs.
// ---------------------------------------------------------------------------
__global__ void __launch_bounds__(512) dense_chain_kernel(
    const float* __restrict__ part, const float* __restrict__ Wd,
    const float* __restrict__ Wcc, const float* __restrict__ bcc,
    const float* __restrict__ Wd2, float* __restrict__ cc) {
    __shared__ float sm[4 * 128];
    __shared__ float sv[4 * 512];
    __shared__ float sz[2 * 256];
    const int tid = threadIdx.x;

    {   // A: finish spatial mean (sum 128 per-row partials, scale by 1/4096)
        float s = 0.0f;
        const float* p = part + (size_t)tid * 128;
        for (int r = 0; r < 128; ++r) s += p[r];
        sm[tid] = s * (1.0f / 4096.0f);
    }
    __syncthreads();

    for (int pr = 0; pr < 4; ++pr) {   // B: sv[pair][o] = Wd @ m[pair]
        float acc = 0.0f;
        const float* w = Wd + (size_t)tid * 128;
        const float* m = sm + pr * 128;
        for (int c = 0; c < 128; ++c) acc += w[c] * m[c];
        sv[pr * 512 + tid] = acc;
    }
    __syncthreads();

    if (tid < 256) {   // C: z[b] = Wcc @ (v_conv0 + v_conv1) + 2*bcc
        for (int bb = 0; bb < 2; ++bb) {
            float acc = 2.0f * bcc[tid];
            const float* w = Wcc + (size_t)tid * 512;
            for (int c = 0; c < 512; ++c)
                acc += w[c] * (sv[bb * 512 + c] + sv[(2 + bb) * 512 + c]);
            sz[bb * 256 + tid] = acc;
        }
    }
    __syncthreads();

    for (int bb = 0; bb < 2; ++bb) {   // D: cc[b][o] = Wd2 @ z[b]
        float acc = 0.0f;
        const float* w = Wd2 + (size_t)tid * 256;
        for (int c = 0; c < 256; ++c) acc += w[c] * sz[bb * 256 + c];
        cc[bb * 512 + tid] = acc;
    }
}

// ---------------------------------------------------------------------------
// out = x + cc[b][c]  (broadcast residual), float4 vectorized
// ---------------------------------------------------------------------------
__global__ void add_residual_kernel(const float4* __restrict__ x4, const float* __restrict__ cc,
                                    float4* __restrict__ out4, int n4) {
    int i = blockIdx.x * blockDim.x + threadIdx.x;
    if (i >= n4) return;
    int e = i * 4;
    int c = (e >> 12) & 511;
    int b = e >> 21;
    float a = cc[b * 512 + c];
    float4 xv = x4[i];
    out4[i] = make_float4(xv.x + a, xv.y + a, xv.z + a, xv.w + a);
}

// ---------------------------------------------------------------------------
extern "C" void kernel_launch(void* const* d_in, const int* in_sizes, int n_in,
                              void* d_out, int out_size, void* d_ws, size_t ws_size,
                              hipStream_t stream) {
    (void)in_sizes; (void)n_in; (void)out_size; (void)ws_size;
    const float* x   = (const float*)d_in[0];
    const float* W2  = (const float*)d_in[3];
    const float* b2  = (const float*)d_in[4];
    const float* W4  = (const float*)d_in[7];
    const float* b4  = (const float*)d_in[8];
    const float* Wd  = (const float*)d_in[9];
    const float* Wcc = (const float*)d_in[10];
    const float* bcc = (const float*)d_in[11];
    const float* Wd2 = (const float*)d_in[12];

    // Workspace layout:
    //   xp   : padded channels-last bf16 x        12,460,032 B
    //   wp   : repacked bf16 weights               2,359,296 B
    //   part : per-row ReLU partial sums             262,144 B
    //   cc   : per-(b,c) channel correction            4,096 B
    char* ws = (char*)d_ws;
    __bf16* xp  = (__bf16*)(ws);
    __bf16* wp  = (__bf16*)(ws + 12460032);
    float* part = (float*)(ws + 12460032 + 2359296);
    float* cc   = (float*)(ws + 12460032 + 2359296 + 262144);

    static bool attr_set = false;   // host-side only; does not affect device work
    if (!attr_set) {
        hipFuncSetAttribute((const void*)conv_wmma_kernel,
                            hipFuncAttributeMaxDynamicSharedMemorySize, LDS_BYTES);
        attr_set = true;
    }

    const int npx = 2 * HP * HP * CIN;               // 6,230,016
    pack_x_kernel<<<(npx + 255) / 256, 256, 0, stream>>>(x, xp, npx);

    const int nw = 2 * 9 * COUT * CIN;               // 1,179,648
    pack_w_kernel<<<(nw + 255) / 256, 256, 0, stream>>>(W2, W4, wp, nw);

    // 256 blocks = (conv, b, row); 8 waves per block
    conv_wmma_kernel<<<256, 256, LDS_BYTES, stream>>>(xp, wp, b2, b4, part);

    dense_chain_kernel<<<1, 512, 0, stream>>>(part, Wd, Wcc, bcc, Wd2, cc);

    const int n4 = 2 * 512 * 64 * 64 / 4;            // 1,048,576
    add_residual_kernel<<<(n4 + 255) / 256, 256, 0, stream>>>((const float4*)x, cc, (float4*)d_out, n4);
}
